// DeltaNet_17162689315496
// MI455X (gfx1250) — compile-verified, matrix operand order checked by reading, not run on hardware
//
#include <hip/hip_runtime.h>
#include <hip/hip_bf16.h>
#include <math.h>

typedef _Float16 f16;
typedef __attribute__((ext_vector_type(16))) _Float16 v16h;
typedef __attribute__((ext_vector_type(8)))  _Float16 v8h;
typedef __attribute__((ext_vector_type(8)))  float    v8f;
typedef __attribute__((ext_vector_type(4)))  int      v4i_;

#define B_   4
#define L_   4096
#define D_   1024
#define H_   4
#define DK_  512
#define DV_  1024
#define DQH  128
#define DVH  256
#define CH   64
#define NCH  (L_/CH)
#define BH   (B_*H_)
#define MROWS (B_*L_)
#define AMLD 68      // fp32 64x64 + pad
#define AMHLD 72     // f16  64x64 + pad

#define WMMA(a,b,c) __builtin_amdgcn_wmma_f32_16x16x32_f16(false,(a),false,(b),(short)0,(c),false,false)

#if defined(__has_builtin)
#if __has_builtin(__builtin_amdgcn_global_load_async_to_lds_b128) && __has_builtin(__builtin_amdgcn_s_wait_asynccnt)
#define USE_ASYNC_LDS 1
#endif
#endif

static __device__ __forceinline__ float sigmoidf_(float x){ return 1.f/(1.f+__expf(-x)); }

static __device__ __forceinline__ v16h pack16(v8h lo, v8h hi){
  v16h r;
#pragma unroll
  for (int t = 0; t < 8; ++t){ r[t] = lo[t]; r[t+8] = hi[t]; }
  return r;
}

static __device__ __forceinline__ v8f vzero(){
  v8f z = {0.f,0.f,0.f,0.f,0.f,0.f,0.f,0.f};
  return z;
}

// contiguous WMMA operand fetch (A-operand row m, or B-operand via [N][K] layout):
// lanes 0-15: K = kbase + {0..7, 16..23}; lanes 16-31: +8
static __device__ __forceinline__ v16h gatherRow(const f16* M, int ld, int row, int kbase, int lane){
  const int k0 = kbase + (((lane >> 4) & 1) << 3);
  v8h lo = *(const v8h*)&M[row*ld + k0];
  v8h hi = *(const v8h*)&M[row*ld + k0 + 16];
  return pack16(lo, hi);
}

// ---------------------------------------------------------------------------
// 1) causal depthwise conv (K=4) + SiLU -> f16
// ---------------------------------------------------------------------------
__global__ __launch_bounds__(256) void conv_silu_k(const float* __restrict__ x,
                                                   const float* __restrict__ w,
                                                   f16* __restrict__ hout){
  const size_t idx = (size_t)blockIdx.x*256 + threadIdx.x;
  const int d = (int)(idx % D_);
  const size_t bl = idx / D_;
  const int l = (int)(bl % L_);
  float acc = 0.f;
#pragma unroll
  for (int i = 0; i < 4; ++i){
    const int ls = l - 3 + i;
    if (ls >= 0) acc += w[d*4+i] * x[(bl - (size_t)l + (size_t)ls)*D_ + d];
  }
  acc = acc * sigmoidf_(acc);
  hout[idx] = (f16)acc;
}

// ---------------------------------------------------------------------------
// 2) weight transpose fp32[K][N] -> f16[N][K]
// ---------------------------------------------------------------------------
__global__ __launch_bounds__(256) void transpose_w(const float* __restrict__ W,
                                                   f16* __restrict__ Wt, int K, int N){
  const size_t idx = (size_t)blockIdx.x*256 + threadIdx.x;
  if (idx >= (size_t)K*N) return;
  const int k = (int)(idx % K);
  const int n = (int)(idx / K);
  Wt[(size_t)n*K + k] = (f16)W[(size_t)k*N + n];
}

// ---------------------------------------------------------------------------
// 3) WMMA GEMM: C[M][N] = A[M][K](f16) * Bt[N][K](f16)^T ; N,K compile-time
// ---------------------------------------------------------------------------
template<int N, int K>
__global__ __launch_bounds__(256) void gemm_f16(const f16* __restrict__ A,
                                                const f16* __restrict__ Bt,
                                                float* __restrict__ C){
  __shared__ f16 As[128][40];   // +8 halves pad -> conflict-free b128 reads
  __shared__ f16 Bs[128][40];
  const int tid = threadIdx.x, lane = tid & 31, w = tid >> 5;
  const int wm = w & 3, wn = w >> 2;
  const int m0 = blockIdx.y * 128, n0 = blockIdx.x * 128;

  v8f acc[2][4];
#pragma unroll
  for (int i = 0; i < 2; ++i)
#pragma unroll
    for (int j = 0; j < 4; ++j) acc[i][j] = vzero();

  const int lrow = tid >> 1, lcol = (tid & 1) * 16;
  const f16* aptr = A + (size_t)(m0 + lrow)*K + lcol;
  const f16* bptr = Bt + (size_t)(n0 + lrow)*K + lcol;
  for (int kk = 0; kk < K; kk += 32){
    const v8h* ap = (const v8h*)(aptr + kk);
    *(v8h*)&As[lrow][lcol]     = ap[0];
    *(v8h*)&As[lrow][lcol + 8] = ap[1];
    const v8h* bp = (const v8h*)(bptr + kk);
    *(v8h*)&Bs[lrow][lcol]     = bp[0];
    *(v8h*)&Bs[lrow][lcol + 8] = bp[1];
    if (kk + 32 < K){
      __builtin_prefetch(aptr + kk + 32, 0, 1);   // -> global_prefetch_b8
      __builtin_prefetch(bptr + kk + 32, 0, 1);
    }
    __syncthreads();

    const int k0 = ((lane >> 4) & 1) * 8;
    v16h af[2], bf[4];
#pragma unroll
    for (int i = 0; i < 2; ++i){
      const int m = wm*32 + i*16 + (lane & 15);
      af[i] = pack16(*(const v8h*)&As[m][k0], *(const v8h*)&As[m][k0 + 16]);
    }
#pragma unroll
    for (int j = 0; j < 4; ++j){
      const int n = wn*64 + j*16 + (lane & 15);
      bf[j] = pack16(*(const v8h*)&Bs[n][k0], *(const v8h*)&Bs[n][k0 + 16]);
    }
#pragma unroll
    for (int i = 0; i < 2; ++i)
#pragma unroll
      for (int j = 0; j < 4; ++j)
        acc[i][j] = WMMA(af[i], bf[j], acc[i][j]);
    __syncthreads();
  }

  float* cbase = C + (size_t)(m0 + wm*32 + ((lane >> 4) << 3))*N + n0 + wn*64 + (lane & 15);
#pragma unroll
  for (int i = 0; i < 2; ++i)
#pragma unroll
    for (int j = 0; j < 4; ++j){
      float* cp = cbase + (size_t)(i*16)*N + j*16;
#pragma unroll
      for (int r = 0; r < 8; ++r)
        cp[(size_t)r*N] = acc[i][j][r];
    }
}

// ---------------------------------------------------------------------------
// 4) per-token prep: beta = sigmoid(h@Wb); l2norm q,k; head-major f16 repack
// ---------------------------------------------------------------------------
__global__ __launch_bounds__(256) void prep_row_k(const f16* __restrict__ hmat,
                                                  const float* __restrict__ Wb,
                                                  const float* __restrict__ qf,
                                                  const float* __restrict__ kf,
                                                  const float* __restrict__ vf,
                                                  f16* __restrict__ qn, f16* __restrict__ kn,
                                                  f16* __restrict__ vh, float* __restrict__ beta){
  const int row = blockIdx.x;
  const int b = row / L_, l = row % L_;
  const int t = threadIdx.x;
  __shared__ float red[8];
  if (t < 8) red[t] = 0.f;
  __syncthreads();
#pragma unroll
  for (int hh = 0; hh < 4; ++hh){
    float p = 0.f;
    for (int d = t; d < D_; d += 256) p += (float)hmat[(size_t)row*D_ + d] * Wb[d*4 + hh];
    atomicAdd(&red[hh], p);
  }
  __syncthreads();
  if (t < 4) beta[((size_t)(b*H_ + t))*L_ + l] = sigmoidf_(red[t]);
  __syncthreads();
  if (t < 8) red[t] = 0.f;
  __syncthreads();
  const float qv0 = qf[(size_t)row*DK_ + t],       qv1 = qf[(size_t)row*DK_ + t + 256];
  const float kv0 = kf[(size_t)row*DK_ + t],       kv1 = kf[(size_t)row*DK_ + t + 256];
  atomicAdd(&red[t >> 7],       qv0*qv0);
  atomicAdd(&red[2 + (t >> 7)], qv1*qv1);
  atomicAdd(&red[4 + (t >> 7)], kv0*kv0);
  atomicAdd(&red[6 + (t >> 7)], kv1*kv1);
  __syncthreads();
  {
    const int h0 = t >> 7, d0 = t & 127;
    const float s0 = 1.f / fmaxf(sqrtf(red[h0]),     1e-12f);
    const float s1 = 1.f / fmaxf(sqrtf(red[2 + h0]), 1e-12f);
    const float u0 = 1.f / fmaxf(sqrtf(red[4 + h0]), 1e-12f);
    const float u1 = 1.f / fmaxf(sqrtf(red[6 + h0]), 1e-12f);
    qn[((size_t)(b*H_ + h0))*L_*DQH     + (size_t)l*DQH + d0] = (f16)(qv0*s0);
    qn[((size_t)(b*H_ + 2 + h0))*L_*DQH + (size_t)l*DQH + d0] = (f16)(qv1*s1);
    kn[((size_t)(b*H_ + h0))*L_*DQH     + (size_t)l*DQH + d0] = (f16)(kv0*u0);
    kn[((size_t)(b*H_ + 2 + h0))*L_*DQH + (size_t)l*DQH + d0] = (f16)(kv1*u1);
  }
  for (int e = t; e < DV_; e += 256){
    const int hh = e >> 8, d0 = e & 255;
    vh[((size_t)(b*H_ + hh))*L_*DVH + (size_t)l*DVH + d0] = (f16)vf[(size_t)row*DV_ + e];
  }
}

// ---------------------------------------------------------------------------
// 5) delta-rule chunked scan. One block per (b,h).
//    St (f16, transposed) in LDS is the WMMA operand view of the state;
//    fp32 master Sg lives in workspace (L2-resident, 2MB total).
//    All WMMA operand fetches are contiguous ds_load_b128 pairs.
// ---------------------------------------------------------------------------
__global__ __launch_bounds__(256) void delta_scan(const f16* __restrict__ qnall,
                                                  const f16* __restrict__ knall,
                                                  const f16* __restrict__ vhall,
                                                  const float* __restrict__ betaall,
                                                  float* __restrict__ Sg,
                                                  float* __restrict__ oout){
  extern __shared__ char smem[];
  float* U   = (float*)smem;                      // [64][256] fp32   65536 B
  float* Am  = U + CH*DVH;                        // [64][68]  fp32   17408 B
  float* bc  = Am + CH*AMLD;                      // [64]               256 B
  f16*  St   = (f16*)(bc + CH);                   // [256][128] f16   65536 B (S^T)
  f16*  Ut   = St + DVH*DQH;                      // [256][64]  f16   32768 B (U^T)
  f16*  Amh  = Ut + DVH*CH;                       // [64][72]   f16    9216 B
  f16*  qs   = Amh + CH*AMHLD;                    // [64][128]  f16   16384 B
  f16*  ks   = qs + CH*DQH;                       // [64][128]  f16   16384 B
  f16*  kt   = ks + CH*DQH;                       // [128][64]  f16   16384 B (k^T)
  f16*  vs   = kt + DQH*CH;                       // [64][256]  f16   32768 B
                                                  // total 272640 B

  const int tid  = threadIdx.x;
  const int lane = tid & 31;
  const int w    = tid >> 5;
  const int bh   = blockIdx.x;
  const int b    = bh >> 2, hh = bh & 3;

  const f16*   qg = qnall  + (size_t)bh*L_*DQH;
  const f16*   kg = knall  + (size_t)bh*L_*DQH;
  const f16*   vg = vhall  + (size_t)bh*L_*DVH;
  const float* bg = betaall + (size_t)bh*L_;
  float*       Sb = Sg + (size_t)bh*DQH*DVH;      // fp32 master state

  for (int i = tid; i < DQH*DVH; i += 256){ Sb[i] = 0.f; }
  for (int i = tid; i < DVH*DQH; i += 256){ St[i] = (f16)0.f; }
  __syncthreads();

  for (int c = 0; c < NCH; ++c){
    const int l0 = c * CH;
    // ---- stage chunk into LDS (ASYNCcnt-tracked DMA when available) ------
#ifdef USE_ASYNC_LDS
    {
      const char* gq = (const char*)(qg + (size_t)l0*DQH);
      const char* gk = (const char*)(kg + (size_t)l0*DQH);
      const char* gv = (const char*)(vg + (size_t)l0*DVH);
      char* lq = (char*)qs;
      char* lk = (char*)ks;
      char* lv = (char*)vs;
#pragma unroll
      for (int t2 = 0; t2 < 4; ++t2){
        const int off = (tid + t2*256) * 16;
        __builtin_amdgcn_global_load_async_to_lds_b128((v4i_*)(gq + off), (v4i_*)(lq + off), 0, 0);
        __builtin_amdgcn_global_load_async_to_lds_b128((v4i_*)(gk + off), (v4i_*)(lk + off), 0, 0);
      }
#pragma unroll
      for (int t2 = 0; t2 < 8; ++t2){
        const int off = (tid + t2*256) * 16;
        __builtin_amdgcn_global_load_async_to_lds_b128((v4i_*)(gv + off), (v4i_*)(lv + off), 0, 0);
      }
      __builtin_amdgcn_s_wait_asynccnt(0);
    }
#else
    for (int i = tid; i < CH*DQH; i += 256){
      qs[i] = qg[(size_t)l0*DQH + i];
      ks[i] = kg[(size_t)l0*DQH + i];
    }
    for (int i = tid; i < CH*DVH; i += 256) vs[i] = vg[(size_t)l0*DVH + i];
#endif
    if (tid < CH) bc[tid] = bg[l0 + tid];
    __syncthreads();

    // build k^T copy (needed by phase 5; completes before the phase-5 barrier)
    for (int i = tid; i < DQH*CH; i += 256){
      const int d = i >> 6, l = i & 63;
      kt[i] = ks[l*DQH + d];
    }

    // ---- phase 1: Am = strict_tril( (beta*k) @ k^T )  (64x64, fp32) -------
    for (int s = w*2; s < w*2 + 2; ++s){
      const int ti = s >> 2, tj = s & 3;
      const int m = ti*16 + (lane & 15);
      const int n = tj*16 + (lane & 15);
      v8f acc = vzero();
      if (ti >= tj){
        const f16 sc = (f16)bc[m];
#pragma unroll
        for (int kb = 0; kb < 4; ++kb){
          v16h a = gatherRow(ks, DQH, m, kb*32, lane);
#pragma unroll
          for (int j2 = 0; j2 < 16; ++j2) a[j2] = a[j2]*sc;
          v16h bb = gatherRow(ks, DQH, n, kb*32, lane);
          acc = WMMA(a, bb, acc);
        }
        if (ti == tj){
          const int mo = ti*16 + ((lane >> 4) << 3);
#pragma unroll
          for (int r = 0; r < 8; ++r) if (mo + r <= n) acc[r] = 0.f;
        }
      }
      const int mo = ti*16 + ((lane >> 4) << 3);
#pragma unroll
      for (int r = 0; r < 8; ++r) Am[(mo + r)*AMLD + n] = acc[r];
    }
    __syncthreads();

    // ---- phase 2: U(=rhs) = beta*v - (beta*k) @ S  (64x256) ---------------
    for (int s = w*8; s < w*8 + 8; ++s){
      const int ti = s >> 4, tj = s & 15;
      const int m  = ti*16 + (lane & 15);
      const int n  = tj*16 + (lane & 15);
      const int mo = ti*16 + ((lane >> 4) << 3);
      v8f acc;
#pragma unroll
      for (int r = 0; r < 8; ++r) acc[r] = bc[mo + r] * (float)vs[(mo + r)*DVH + n];
      const f16 sc = (f16)(-bc[m]);
#pragma unroll
      for (int kb = 0; kb < 4; ++kb){
        v16h a = gatherRow(ks, DQH, m, kb*32, lane);
#pragma unroll
        for (int j2 = 0; j2 < 16; ++j2) a[j2] = a[j2]*sc;
        v16h bb = gatherRow(St, DQH, n, kb*32, lane);   // b[j] = S[k][n] = St[n][k]
        acc = WMMA(a, bb, acc);
      }
#pragma unroll
      for (int r = 0; r < 8; ++r) U[(mo + r)*DVH + n] = acc[r];
    }
    __syncthreads();

    // ---- phase 3: forward substitution (I+A)U = rhs; emit U^T (f16) -------
    {
      const int n = tid;
      Ut[n*CH] = (f16)U[n];
      for (int i = 1; i < CH; ++i){
        float a2 = U[i*DVH + n];
        const float* arow = &Am[i*AMLD];
        for (int j = 0; j < i; ++j) a2 -= arow[j] * U[j*DVH + n];
        U[i*DVH + n] = a2;
        Ut[n*CH + i] = (f16)a2;
      }
    }
    __syncthreads();

    // ---- phase 4a: Amh = tril( q @ k^T )  (incl. diagonal, f16) -----------
    for (int s = w*2; s < w*2 + 2; ++s){
      const int ti = s >> 2, tj = s & 3;
      const int m = ti*16 + (lane & 15);
      const int n = tj*16 + (lane & 15);
      v8f acc = vzero();
      if (ti >= tj){
#pragma unroll
        for (int kb = 0; kb < 4; ++kb){
          v16h a  = gatherRow(qs, DQH, m, kb*32, lane);
          v16h bb = gatherRow(ks, DQH, n, kb*32, lane);
          acc = WMMA(a, bb, acc);
        }
        if (ti == tj){
          const int mo = ti*16 + ((lane >> 4) << 3);
#pragma unroll
          for (int r = 0; r < 8; ++r) if (mo + r < n) acc[r] = 0.f;
        }
      }
      const int mo = ti*16 + ((lane >> 4) << 3);
#pragma unroll
      for (int r = 0; r < 8; ++r) Amh[(mo + r)*AMHLD + n] = (f16)acc[r];
    }
    __syncthreads();

    // ---- phase 4b: o = q @ S + Amh @ U  (64x256) -> global ----------------
    for (int s = w*8; s < w*8 + 8; ++s){
      const int ti = s >> 4, tj = s & 15;
      const int m = ti*16 + (lane & 15);
      const int n = tj*16 + (lane & 15);
      v8f acc = vzero();
#pragma unroll
      for (int kb = 0; kb < 4; ++kb){
        v16h a  = gatherRow(qs, DQH, m, kb*32, lane);
        v16h bb = gatherRow(St, DQH, n, kb*32, lane);
        acc = WMMA(a, bb, acc);
      }
#pragma unroll
      for (int kb = 0; kb < 2; ++kb){
        v16h a  = gatherRow(Amh, AMHLD, m, kb*32, lane);
        v16h bb = gatherRow(Ut, CH, n, kb*32, lane);    // b[j] = U[k][n] = Ut[n][k]
        acc = WMMA(a, bb, acc);
      }
      const int mo = ti*16 + ((lane >> 4) << 3);
#pragma unroll
      for (int r = 0; r < 8; ++r){
        const int l = l0 + mo + r;
        oout[((size_t)(b*L_ + l)*H_ + hh)*DVH + n] = acc[r];
      }
    }
    __syncthreads();

    // ---- phase 5: S += k^T @ U ; fp32 master in Sg (L2), f16 view in St ---
    {
      const int s0 = w*16;
      for (int s = s0; s < s0 + 16; ++s){
        const int ti = s >> 4, tj = s & 15;   // ti == w
        const int m  = ti*16 + (lane & 15);
        const int n  = tj*16 + (lane & 15);
        const int mo = ti*16 + ((lane >> 4) << 3);
        v8f acc;
#pragma unroll
        for (int r = 0; r < 8; ++r) acc[r] = Sb[(mo + r)*DVH + n];
#pragma unroll
        for (int kb = 0; kb < 2; ++kb){
          v16h a  = gatherRow(kt, CH, m, kb*32, lane);  // a[j] = k[k][m] = kt[m][k]
          v16h bb = gatherRow(Ut, CH, n, kb*32, lane);
          acc = WMMA(a, bb, acc);
        }
        v8h hv;
#pragma unroll
        for (int r = 0; r < 8; ++r){
          Sb[(mo + r)*DVH + n] = acc[r];
          hv[r] = (f16)acc[r];
        }
        *(v8h*)&St[n*DQH + mo] = hv;                    // transposed f16 view
      }
    }
    __syncthreads();
  }
}

// ---------------------------------------------------------------------------
// 6) fused RMSNorm * norm_w * silu(g) -> f16
// ---------------------------------------------------------------------------
__global__ __launch_bounds__(256) void gate_norm_k(const float* __restrict__ o,
                                                   const float* __restrict__ g,
                                                   const float* __restrict__ nw,
                                                   f16* __restrict__ gated){
  const int row = blockIdx.x;
  const int t = threadIdx.x;
  __shared__ float ss[4];
  if (t < 4) ss[t] = 0.f;
  __syncthreads();
  float ov[4];
#pragma unroll
  for (int hh = 0; hh < 4; ++hh){
    ov[hh] = o[(size_t)row*DV_ + hh*DVH + t];
    atomicAdd(&ss[hh], ov[hh]*ov[hh]);
  }
  __syncthreads();
  const float wv = nw[t];
#pragma unroll
  for (int hh = 0; hh < 4; ++hh){
    const float r  = rsqrtf(ss[hh]*(1.f/DVH) + 1e-5f);
    const float gv = g[(size_t)row*DV_ + hh*DVH + t];
    gated[(size_t)row*DV_ + hh*DVH + t] = (f16)(ov[hh]*r*wv*(gv*sigmoidf_(gv)));
  }
}

// ---------------------------------------------------------------------------
extern "C" void kernel_launch(void* const* d_in, const int* in_sizes, int n_in,
                              void* d_out, int out_size, void* d_ws, size_t ws_size,
                              hipStream_t stream){
  const float* x     = (const float*)d_in[0];
  const float* convw = (const float*)d_in[1];
  const float* Wq    = (const float*)d_in[2];
  const float* Wk    = (const float*)d_in[3];
  const float* Wv    = (const float*)d_in[4];
  const float* Wb    = (const float*)d_in[5];
  const float* Wg    = (const float*)d_in[6];
  const float* nw    = (const float*)d_in[7];
  const float* Wo    = (const float*)d_in[8];
  float* out = (float*)d_out;
  (void)in_sizes; (void)n_in; (void)out_size; (void)ws_size;

  char* ws = (char*)d_ws;
  size_t off = 0;
  auto carve = [&](size_t bytes)->char*{
    char* p = ws + off;
    off += (bytes + 255) & ~(size_t)255;
    return p;
  };
  f16*   hf    = (f16*)  carve((size_t)MROWS*D_*2);
  f16*   WqT   = (f16*)  carve((size_t)DK_*D_*2);
  f16*   WkT   = (f16*)  carve((size_t)DK_*D_*2);
  f16*   WvT   = (f16*)  carve((size_t)DV_*D_*2);
  f16*   WgT   = (f16*)  carve((size_t)DV_*D_*2);
  f16*   WoT   = (f16*)  carve((size_t)D_*DV_*2);
  float* qf    = (float*)carve((size_t)MROWS*DK_*4);
  float* kf    = (float*)carve((size_t)MROWS*DK_*4);
  float* vf    = (float*)carve((size_t)MROWS*DV_*4);
  float* gf    = (float*)carve((size_t)MROWS*DV_*4);
  f16*   qn    = (f16*)  carve((size_t)BH*L_*DQH*2);
  f16*   kn    = (f16*)  carve((size_t)BH*L_*DQH*2);
  f16*   vh16  = (f16*)  carve((size_t)BH*L_*DVH*2);
  float* beta  = (float*)carve((size_t)BH*L_*4);
  float* Sg    = (float*)carve((size_t)BH*DQH*DVH*4);   // 2MB fp32 state (L2)
  float* ofs   = (float*)carve((size_t)MROWS*DV_*4);
  f16*   gated = (f16*)  carve((size_t)MROWS*DV_*2);

  // 1) conv + silu
  conv_silu_k<<<(MROWS*D_)/256, 256, 0, stream>>>(x, convw, hf);

  // 2) weight transposes (fp32 -> f16 [N][K])
  transpose_w<<<(D_*DK_ + 255)/256, 256, 0, stream>>>(Wq, WqT, D_, DK_);
  transpose_w<<<(D_*DK_ + 255)/256, 256, 0, stream>>>(Wk, WkT, D_, DK_);
  transpose_w<<<(D_*DV_ + 255)/256, 256, 0, stream>>>(Wv, WvT, D_, DV_);
  transpose_w<<<(D_*DV_ + 255)/256, 256, 0, stream>>>(Wg, WgT, D_, DV_);
  transpose_w<<<(DV_*D_ + 255)/256, 256, 0, stream>>>(Wo, WoT, DV_, D_);

  // 3) projection GEMMs (WMMA)
  gemm_f16<DK_, D_><<<dim3(DK_/128, MROWS/128), 256, 0, stream>>>(hf, WqT, qf);
  gemm_f16<DK_, D_><<<dim3(DK_/128, MROWS/128), 256, 0, stream>>>(hf, WkT, kf);
  gemm_f16<DV_, D_><<<dim3(DV_/128, MROWS/128), 256, 0, stream>>>(hf, WvT, vf);
  gemm_f16<DV_, D_><<<dim3(DV_/128, MROWS/128), 256, 0, stream>>>(hf, WgT, gf);

  // 4) beta / l2norm / head-major repack
  prep_row_k<<<MROWS, 256, 0, stream>>>(hf, Wb, qf, kf, vf, qn, kn, vh16, beta);

  // 5) delta-rule scan (WMMA) — one block per (b,h)
  const size_t scan_lds = (size_t)(CH*DVH + CH*AMLD + CH)*4
                        + (size_t)(DVH*DQH + DVH*CH + CH*AMHLD + 2*CH*DQH + DQH*CH + CH*DVH)*2; // 272640 B
  delta_scan<<<BH, 256, scan_lds, stream>>>(qn, kn, vh16, beta, Sg, ofs);

  // 6) gated RMS norm
  gate_norm_k<<<MROWS, 256, 0, stream>>>(ofs, gf, nw, gated);

  // 7) output projection (WMMA) -> fp32 out
  gemm_f16<D_, DV_><<<dim3(D_/128, MROWS/128), 256, 0, stream>>>(gated, WoT, out);
}
